// AdaptiveRecursiveRPC_91199335563703
// MI455X (gfx1250) — matrix operand order repaired and checked by reading, not compiled, not run on hardware
//
#include <hip/hip_runtime.h>

#define D_MODEL 1024
#define NUM_EXPERTS 16
#define BATCH 8192
#define LN_EPS 1e-5f

typedef __attribute__((ext_vector_type(16))) __bf16 v16bf;
typedef __attribute__((ext_vector_type(8)))  float v8f;

struct alignas(16) U128 { unsigned x, y, z, w; };

union Frag {
    U128  u[2];
    v16bf v;
};

// ---------------------------------------------------------------------------
// gfx1250 async global->LDS path (ASYNCcnt-tracked, no VGPR staging).
// Builtin params are (v4i addrspace(1)*, v4i addrspace(3)*, imm, imm).
// ---------------------------------------------------------------------------
#if defined(__AMDGCN__) && __has_builtin(__builtin_amdgcn_global_load_async_to_lds_b128) && \
    __has_builtin(__builtin_amdgcn_s_wait_asynccnt)
#define USE_ASYNC 1
typedef int v4i __attribute__((vector_size(16)));
typedef __attribute__((address_space(1))) v4i* g_v4i_p;
typedef __attribute__((address_space(3))) v4i* l_v4i_p;
#else
#define USE_ASYNC 0
#endif

__device__ __forceinline__ void copy16(unsigned short* lds, const unsigned short* g) {
#if USE_ASYNC
    __builtin_amdgcn_global_load_async_to_lds_b128(
        (g_v4i_p)(size_t)g, (l_v4i_p)lds, 0, 0);
#else
    *(U128*)lds = *(const U128*)g;
#endif
}

__device__ __forceinline__ void wait_async() {
#if USE_ASYNC
    __builtin_amdgcn_s_wait_asynccnt(0);
#endif
}

__device__ __forceinline__ unsigned short f2bf(float f) {
    unsigned u = __builtin_bit_cast(unsigned, f);
    unsigned r = u + 0x7FFFu + ((u >> 16) & 1u);   // round-to-nearest-even
    return (unsigned short)(r >> 16);
}

// ---------------------------------------------------------------------------
// K1: f32 -> bf16 conversion of h, vectorized 4 elems/thread
// ---------------------------------------------------------------------------
__global__ __launch_bounds__(256)
void convert_h_kernel(const float* __restrict__ h, unsigned short* __restrict__ hbf) {
    size_t i = ((size_t)blockIdx.x * 256 + threadIdx.x) * 4;
    float4 v = *(const float4*)(h + i);
    unsigned lo = (unsigned)f2bf(v.x) | ((unsigned)f2bf(v.y) << 16);
    unsigned hi = (unsigned)f2bf(v.z) | ((unsigned)f2bf(v.w) << 16);
    uint2 o; o.x = lo; o.y = hi;
    *(uint2*)(hbf + i) = o;
}

// ---------------------------------------------------------------------------
// K2: transpose + convert a [1024(d)][1024(k)] f32 matrix into bf16 [k][d].
// ---------------------------------------------------------------------------
__global__ __launch_bounds__(256)
void transpose_convert_kernel(const float* __restrict__ src, unsigned short* __restrict__ dst) {
    __shared__ unsigned short tile[64][65];
    const float* s = src + (size_t)blockIdx.z * D_MODEL * D_MODEL;
    unsigned short* d = dst + (size_t)blockIdx.z * D_MODEL * D_MODEL;
    int d0 = blockIdx.y * 64, k0 = blockIdx.x * 64;
    int t = threadIdx.x;
    #pragma unroll
    for (int i = 0; i < 16; ++i) {
        int idx = t + i * 256;
        int r = idx >> 6, c = idx & 63;
        tile[r][c] = f2bf(s[(size_t)(d0 + r) * D_MODEL + k0 + c]);
    }
    __syncthreads();
    #pragma unroll
    for (int i = 0; i < 16; ++i) {
        int idx = t + i * 256;
        int kk = idx >> 6, dd = idx & 63;
        d[(size_t)(k0 + kk) * D_MODEL + d0 + dd] = tile[dd][kk];
    }
}

// ---------------------------------------------------------------------------
// LDS staging + WMMA fragment helpers (wave32 layouts per CDNA5 ISA 7.12.2)
// Both A and B tiles are 128 rows x 32 K bf16, LDS row stride 40 (80B, 16B ok)
// ---------------------------------------------------------------------------
#define LDS_STRIDE 40
#define TILE_WORDS (128 * LDS_STRIDE)

// Stage one 128x32 A tile and one 128x32 B tile: 512 16B chunks each,
// 2 chunks/thread/tile, straight into LDS via async DMA when available.
__device__ __forceinline__ void stage_tile(const unsigned short* __restrict__ gA,
                                           const unsigned short* __restrict__ gB,
                                           unsigned short* ldsA, unsigned short* ldsB,
                                           int t) {
    int c0 = t, c1 = t + 256;
    copy16(ldsA + (c0 >> 2) * LDS_STRIDE + (c0 & 3) * 8,
           gA + (size_t)(c0 >> 2) * D_MODEL + (c0 & 3) * 8);
    copy16(ldsA + (c1 >> 2) * LDS_STRIDE + (c1 & 3) * 8,
           gA + (size_t)(c1 >> 2) * D_MODEL + (c1 & 3) * 8);
    copy16(ldsB + (c0 >> 2) * LDS_STRIDE + (c0 & 3) * 8,
           gB + (size_t)(c0 >> 2) * D_MODEL + (c0 & 3) * 8);
    copy16(ldsB + (c1 >> 2) * LDS_STRIDE + (c1 & 3) * 8,
           gB + (size_t)(c1 >> 2) * D_MODEL + (c1 & 3) * 8);
}

// A fragment, 16x32 bf16: lanes 0-15 = rows M, halves pick K octets.
__device__ __forceinline__ v16bf load_fragA(const unsigned short* ldsA, int mrow, int lane) {
    int row = mrow + (lane & 15);
    int kb  = (lane >> 4) * 8;
    Frag f;
    f.u[0] = *(const U128*)(ldsA + row * LDS_STRIDE + kb);
    f.u[1] = *(const U128*)(ldsA + row * LDS_STRIDE + 16 + kb);
    return f.v;
}

// B fragment, 32x16 bf16: lane = column N, halves pick K octets.
__device__ __forceinline__ v16bf load_fragB(const unsigned short* ldsB, int ncol, int lane) {
    int col = ncol + (lane & 15);
    int kb  = (lane >> 4) * 8;
    Frag f;
    f.u[0] = *(const U128*)(ldsB + col * LDS_STRIDE + kb);
    f.u[1] = *(const U128*)(ldsB + col * LDS_STRIDE + kb + 16);
    return f.v;
}

#define WMMA_BF16(a, b, c) \
    __builtin_amdgcn_wmma_f32_16x16x32_bf16(false, (a), false, (b), (short)0, (c), false, false)

// ---------------------------------------------------------------------------
// K3: router GEMM  hidden = silu(h @ w1 + b1).  Block tile 128x128, 8 waves
// as 4(M)x2(N), each wave owns 32x64 = 8 WMMA tiles.  Double-buffered LDS.
// ---------------------------------------------------------------------------
__global__ __launch_bounds__(256)
void router_gemm_kernel(const unsigned short* __restrict__ hbf,
                        const unsigned short* __restrict__ w1bf,   // [k][d] bf16
                        const float* __restrict__ b1,
                        float* __restrict__ hidden) {
    __shared__ unsigned short ldsA[2][TILE_WORDS];
    __shared__ unsigned short ldsB[2][TILE_WORDS];
    int t = threadIdx.x;
    int lane = t & 31, wave = t >> 5;
    int wm = wave & 3, wn = wave >> 2;
    int m0 = blockIdx.y * 128;
    int n0 = blockIdx.x * 128;

    const unsigned short* gA = hbf + (size_t)m0 * D_MODEL;
    const unsigned short* gB = w1bf + (size_t)n0 * D_MODEL;

    v8f acc[2][4] = {};

    stage_tile(gA, gB, ldsA[0], ldsB[0], t);
    wait_async();
    __syncthreads();

    int cur = 0;
    for (int kt = 0; kt < D_MODEL; kt += 32) {
        int nxt = cur ^ 1;
        if (kt + 32 < D_MODEL)
            stage_tile(gA + kt + 32, gB + kt + 32, ldsA[nxt], ldsB[nxt], t);

        v16bf a0 = load_fragA(ldsA[cur], wm * 32, lane);
        v16bf a1 = load_fragA(ldsA[cur], wm * 32 + 16, lane);
        #pragma unroll
        for (int j = 0; j < 4; ++j) {
            v16bf b = load_fragB(ldsB[cur], wn * 64 + j * 16, lane);
            acc[0][j] = WMMA_BF16(a0, b, acc[0][j]);
            acc[1][j] = WMMA_BF16(a1, b, acc[1][j]);
        }
        wait_async();
        __syncthreads();
        cur = nxt;
    }

    #pragma unroll
    for (int i = 0; i < 2; ++i) {
        int rbase = m0 + wm * 32 + i * 16 + ((lane >> 4) << 3);
        #pragma unroll
        for (int j = 0; j < 4; ++j) {
            int col = n0 + wn * 64 + j * 16 + (lane & 15);
            float bias = b1[col];
            #pragma unroll
            for (int v = 0; v < 8; ++v) {
                float x = acc[i][j][v] + bias;
                float sig = 1.0f / (1.0f + __expf(-x));
                hidden[(size_t)(rbase + v) * D_MODEL + col] = x * sig;
            }
        }
    }
}

// ---------------------------------------------------------------------------
// K4: logits = hidden @ w2 + b2 ; weights = softmax(logits).  One block/row.
// ---------------------------------------------------------------------------
__global__ __launch_bounds__(128)
void router_logits_kernel(const float* __restrict__ hidden,
                          const float* __restrict__ w2,  // [d][16]
                          const float* __restrict__ b2,
                          float* __restrict__ logits,
                          float* __restrict__ weights) {
    __shared__ float lg[16];
    int row = blockIdx.x;
    int t = threadIdx.x;
    int e = t >> 3, sub = t & 7;
    const float* hr = hidden + (size_t)row * D_MODEL;

    float s = 0.f;
    for (int d = sub; d < D_MODEL; d += 8)
        s += hr[d] * w2[(size_t)d * NUM_EXPERTS + e];
    s += __shfl_xor(s, 1);
    s += __shfl_xor(s, 2);
    s += __shfl_xor(s, 4);
    if (sub == 0) {
        float l = s + b2[e];
        lg[e] = l;
        logits[(size_t)row * NUM_EXPERTS + e] = l;
    }
    __syncthreads();
    if (t < 32) {
        float x = (t < 16) ? lg[t] : -1e30f;
        float m = x;
        m = fmaxf(m, __shfl_xor(m, 1));
        m = fmaxf(m, __shfl_xor(m, 2));
        m = fmaxf(m, __shfl_xor(m, 4));
        m = fmaxf(m, __shfl_xor(m, 8));
        float ex = (t < 16) ? __expf(x - m) : 0.f;
        float sum = ex;
        sum += __shfl_xor(sum, 1);
        sum += __shfl_xor(sum, 2);
        sum += __shfl_xor(sum, 4);
        sum += __shfl_xor(sum, 8);
        if (t < 16) weights[(size_t)row * NUM_EXPERTS + t] = ex / sum;
    }
}

// ---------------------------------------------------------------------------
// K5: combined[b,k] = sum_e weights[b,e] * (h @ experts[e])[b,k]
// Flat 512-step (16 experts x 32 k-steps) double-buffered pipeline so the
// async prefetch flows across expert boundaries; per-expert WMMA results are
// scaled per-row by weights[b,e] and folded into register accumulators.
// ---------------------------------------------------------------------------
__global__ __launch_bounds__(256)
void expert_combine_kernel(const unsigned short* __restrict__ hbf,
                           const unsigned short* __restrict__ ebf,  // [e][k][d] bf16
                           const float* __restrict__ weights,       // [B][16]
                           float* __restrict__ combined) {
    __shared__ unsigned short ldsA[2][TILE_WORDS];
    __shared__ unsigned short ldsB[2][TILE_WORDS];
    int t = threadIdx.x;
    int lane = t & 31, wave = t >> 5;
    int wm = wave & 3, wn = wave >> 2;
    int m0 = blockIdx.y * 128;
    int n0 = blockIdx.x * 128;

    const unsigned short* gA = hbf + (size_t)m0 * D_MODEL;

    v8f comb[2][4] = {};
    v8f acc[2][4] = {};

    stage_tile(gA, ebf + (size_t)n0 * D_MODEL, ldsA[0], ldsB[0], t);
    wait_async();
    __syncthreads();

    int cur = 0;
    for (int s = 0; s < NUM_EXPERTS * 32; ++s) {
        int nxt = cur ^ 1;
        if (s + 1 < NUM_EXPERTS * 32) {
            int e1 = (s + 1) >> 5;
            int kt1 = ((s + 1) & 31) << 5;
            stage_tile(gA + kt1,
                       ebf + ((size_t)e1 * D_MODEL + n0) * D_MODEL + kt1,
                       ldsA[nxt], ldsB[nxt], t);
        }

        v16bf a0 = load_fragA(ldsA[cur], wm * 32, lane);
        v16bf a1 = load_fragA(ldsA[cur], wm * 32 + 16, lane);
        #pragma unroll
        for (int j = 0; j < 4; ++j) {
            v16bf b = load_fragB(ldsB[cur], wn * 64 + j * 16, lane);
            acc[0][j] = WMMA_BF16(a0, b, acc[0][j]);
            acc[1][j] = WMMA_BF16(a1, b, acc[1][j]);
        }

        if ((s & 31) == 31) {                      // end of this expert
            int e = s >> 5;
            #pragma unroll
            for (int i = 0; i < 2; ++i) {
                int rbase = m0 + wm * 32 + i * 16 + ((lane >> 4) << 3);
                #pragma unroll
                for (int v = 0; v < 8; ++v) {
                    float wv = weights[(size_t)(rbase + v) * NUM_EXPERTS + e];
                    #pragma unroll
                    for (int j = 0; j < 4; ++j) {
                        comb[i][j][v] += wv * acc[i][j][v];
                        acc[i][j][v] = 0.f;
                    }
                }
            }
        }
        wait_async();
        __syncthreads();
        cur = nxt;
    }

    #pragma unroll
    for (int i = 0; i < 2; ++i) {
        int rbase = m0 + wm * 32 + i * 16 + ((lane >> 4) << 3);
        #pragma unroll
        for (int j = 0; j < 4; ++j) {
            int col = n0 + wn * 64 + j * 16 + (lane & 15);
            #pragma unroll
            for (int v = 0; v < 8; ++v)
                combined[(size_t)(rbase + v) * D_MODEL + col] = comb[i][j][v];
        }
    }
}

// ---------------------------------------------------------------------------
// K6: h_next = LayerNorm(h + combined) * gamma + beta.  One block per row.
// ---------------------------------------------------------------------------
__global__ __launch_bounds__(256)
void residual_ln_kernel(const float* __restrict__ h,
                        const float* __restrict__ combined,
                        const float* __restrict__ gamma,
                        const float* __restrict__ beta,
                        float* __restrict__ out) {
    __shared__ float ss[8], qq[8];
    int row = blockIdx.x;
    int t = threadIdx.x;
    int lane = t & 31, wave = t >> 5;
    size_t base = (size_t)row * D_MODEL + t * 4;

    float4 a = *(const float4*)(h + base);
    float4 c = *(const float4*)(combined + base);
    float4 x;
    x.x = a.x + c.x; x.y = a.y + c.y; x.z = a.z + c.z; x.w = a.w + c.w;

    float s = x.x + x.y + x.z + x.w;
    float q = x.x * x.x + x.y * x.y + x.z * x.z + x.w * x.w;
    #pragma unroll
    for (int k = 1; k < 32; k <<= 1) {
        s += __shfl_xor(s, k);
        q += __shfl_xor(q, k);
    }
    if (lane == 0) { ss[wave] = s; qq[wave] = q; }
    __syncthreads();
    if (t == 0) {
        float S = 0.f, Q = 0.f;
        #pragma unroll
        for (int i = 0; i < 8; ++i) { S += ss[i]; Q += qq[i]; }
        float mean = S * (1.0f / D_MODEL);
        float var  = Q * (1.0f / D_MODEL) - mean * mean;
        ss[0] = mean;
        qq[0] = rsqrtf(var + LN_EPS);
    }
    __syncthreads();
    float mean = ss[0], inv = qq[0];

    float4 g = *(const float4*)(gamma + t * 4);
    float4 bt = *(const float4*)(beta + t * 4);
    float4 o;
    o.x = (x.x - mean) * inv * g.x + bt.x;
    o.y = (x.y - mean) * inv * g.y + bt.y;
    o.z = (x.z - mean) * inv * g.z + bt.z;
    o.w = (x.w - mean) * inv * g.w + bt.w;
    *(float4*)(out + base) = o;
}

// ---------------------------------------------------------------------------
extern "C" void kernel_launch(void* const* d_in, const int* in_sizes, int n_in,
                              void* d_out, int out_size, void* d_ws, size_t ws_size,
                              hipStream_t stream) {
    const float* h       = (const float*)d_in[0];
    const float* w1      = (const float*)d_in[1];
    const float* b1      = (const float*)d_in[2];
    const float* w2      = (const float*)d_in[3];
    const float* b2      = (const float*)d_in[4];
    const float* experts = (const float*)d_in[5];
    const float* gamma   = (const float*)d_in[6];
    const float* beta    = (const float*)d_in[7];

    float* out_h = (float*)d_out;
    float* out_w = out_h + (size_t)BATCH * D_MODEL;
    float* out_l = out_w + (size_t)BATCH * NUM_EXPERTS;

    char* ws = (char*)d_ws;
    unsigned short* hbf  = (unsigned short*)ws; ws += (size_t)BATCH * D_MODEL * 2;
    unsigned short* w1bf = (unsigned short*)ws; ws += (size_t)D_MODEL * D_MODEL * 2;
    unsigned short* ebf  = (unsigned short*)ws; ws += (size_t)NUM_EXPERTS * D_MODEL * D_MODEL * 2;
    float* hidden   = (float*)ws; ws += (size_t)BATCH * D_MODEL * 4;
    float* combined = (float*)ws;

    // 1) h -> bf16
    convert_h_kernel<<<(BATCH * D_MODEL) / (256 * 4), 256, 0, stream>>>(h, hbf);
    // 2) w1, experts -> bf16 transposed [k][d]
    transpose_convert_kernel<<<dim3(16, 16, 1), 256, 0, stream>>>(w1, w1bf);
    transpose_convert_kernel<<<dim3(16, 16, NUM_EXPERTS), 256, 0, stream>>>(experts, ebf);
    // 3) hidden = silu(h @ w1 + b1)
    router_gemm_kernel<<<dim3(D_MODEL / 128, BATCH / 128), 256, 0, stream>>>(hbf, w1bf, b1, hidden);
    // 4) logits + softmax -> d_out slices (weights feeds K5)
    router_logits_kernel<<<BATCH, 128, 0, stream>>>(hidden, w2, b2, out_l, out_w);
    // 5) combined = sum_e weights[:,e] * (h @ experts[e])
    expert_combine_kernel<<<dim3(D_MODEL / 128, BATCH / 128), 256, 0, stream>>>(hbf, ebf, out_w, combined);
    // 6) h_next = LN(h + combined)
    residual_ln_kernel<<<BATCH, 256, 0, stream>>>(h, combined, gamma, beta, out_h);
}
